// DualAttention_42958262894877
// MI455X (gfx1250) — matrix operand order repaired and checked by reading, not compiled, not run on hardware
//
#include <hip/hip_runtime.h>
#include <hip/hip_fp16.h>
#include <hip/hip_bf16.h>

typedef __attribute__((ext_vector_type(16))) _Float16 v16h;
typedef __attribute__((ext_vector_type(8)))  _Float16 v8h;
typedef __attribute__((ext_vector_type(8)))  float    v8f;
typedef int v4i __attribute__((vector_size(16)));

#define B_    8
#define L_    8
#define C_    64
#define W_    32
#define WH    1024
#define LWH   8192
#define NH_   128
#define LN_EPS 1e-5f

__device__ __forceinline__ v8f wmma_f16(v16h a, v16h b, v8f c) {
  return __builtin_amdgcn_wmma_f32_16x16x32_f16(
      false, a, false, b, (short)0, c, false, false);
}

// ---- gfx1250 async global->LDS copy (ASYNCcnt-tracked DMA) ----------------
__device__ __forceinline__ void async_ld16(const _Float16* g, _Float16* l) {
#if __has_builtin(__builtin_amdgcn_global_load_async_to_lds_b128)
  __builtin_amdgcn_global_load_async_to_lds_b128(
      (__attribute__((address_space(1))) v4i*)g,
      (__attribute__((address_space(3))) v4i*)l, 0, 0);
#else
  asm volatile("global_load_async_to_lds_b128 %0, %1, off"
               :: "v"((unsigned)(unsigned long long)l),
                  "v"((unsigned long long)g)
               : "memory");
#endif
}
__device__ __forceinline__ void wait_async0() {
#if __has_builtin(__builtin_amdgcn_s_wait_asynccnt)
  __builtin_amdgcn_s_wait_asynccnt(0);
#else
  asm volatile("s_wait_asynccnt 0x0" ::: "memory");
#endif
}

// ---- DPP 16-lane butterfly reductions (VALU, no LDS traffic) --------------
template <int CTRL>
__device__ __forceinline__ float dpp_f(float x) {
  return __int_as_float(__builtin_amdgcn_update_dpp(
      0, __float_as_int(x), CTRL, 0xF, 0xF, true));
}
__device__ __forceinline__ float red_max16(float v) {
  v = fmaxf(v, dpp_f<0xB1>(v));
  v = fmaxf(v, dpp_f<0x4E>(v));
  v = fmaxf(v, dpp_f<0x141>(v));
  v = fmaxf(v, dpp_f<0x140>(v));
  return v;
}
__device__ __forceinline__ float red_sum16(float v) {
  v += dpp_f<0xB1>(v);
  v += dpp_f<0x4E>(v);
  v += dpp_f<0x141>(v);
  v += dpp_f<0x140>(v);
  return v;
}

// ---- fragment build helpers ----------------------------------------------
__device__ __forceinline__ v16h ld16_split(const _Float16* p) {
  v8h lo = *(const v8h*)p;
  v8h hi = *(const v8h*)(p + 16);
  return __builtin_shufflevector(lo, hi, 0,1,2,3,4,5,6,7,8,9,10,11,12,13,14,15);
}
__device__ __forceinline__ v16h ld16_cont(const _Float16* p) {
  v8h lo = *(const v8h*)p;
  v8h hi = *(const v8h*)(p + 8);
  return __builtin_shufflevector(lo, hi, 0,1,2,3,4,5,6,7,8,9,10,11,12,13,14,15);
}
__device__ __forceinline__ v16h cvt16(float4 a, float4 b, float4 c, float4 d) {
  v16h r;
  r[0]=(_Float16)a.x;  r[1]=(_Float16)a.y;  r[2]=(_Float16)a.z;  r[3]=(_Float16)a.w;
  r[4]=(_Float16)b.x;  r[5]=(_Float16)b.y;  r[6]=(_Float16)b.z;  r[7]=(_Float16)b.w;
  r[8]=(_Float16)c.x;  r[9]=(_Float16)c.y;  r[10]=(_Float16)c.z; r[11]=(_Float16)c.w;
  r[12]=(_Float16)d.x; r[13]=(_Float16)d.y; r[14]=(_Float16)d.z; r[15]=(_Float16)d.w;
  return r;
}

// ---------------------------------------------------------------------------
// f16 pre-conversion of K (transposed to [b][key][c]) and V ([b][c][key]).
// ---------------------------------------------------------------------------
__global__ __launch_bounds__(256) void khconv_kernel(
    const float* __restrict__ keys, _Float16* __restrict__ kh)
{
  const int i = blockIdx.x * 256 + threadIdx.x;     // over B*LWH rows
  const int b = i >> 13, kg = i & 8191;
  const float* src = keys + ((size_t)b * L_ * C_ + (kg >> 10) * C_) * WH + (kg & 1023);
  _Float16* dst = kh + (size_t)i * 64;
  #pragma unroll 8
  for (int c = 0; c < 64; ++c) dst[c] = (_Float16)src[c * WH];
}
__global__ __launch_bounds__(256) void vhconv_kernel(
    const float* __restrict__ v, _Float16* __restrict__ vh)
{
  const size_t i = ((size_t)blockIdx.x * 256 + threadIdx.x) * 4;
  const float4 x = *(const float4*)(v + i);
  _Float16* d = vh + i;
  d[0] = (_Float16)x.x; d[1] = (_Float16)x.y;
  d[2] = (_Float16)x.z; d[3] = (_Float16)x.w;
}

// ---------------------------------------------------------------------------
// Spatial attention, flash style, double-buffered async LDS staging.
// Grid: B*16 blocks, 128 threads (4 waves); KT=64 keys/iter.
// ---------------------------------------------------------------------------
__global__ __launch_bounds__(128) void spatial_attn_kernel(
    const float* __restrict__ q, const _Float16* __restrict__ kh,
    const _Float16* __restrict__ vh, float* __restrict__ res_sp)
{
  __shared__ __align__(32) char     smemA[2 * 64 * 72 * 2];  // Kt dbl-buf / Ost
  __shared__ __align__(32) _Float16 VtB[2][64][72];
  __shared__ __align__(32) _Float16 Pst[4][16][72];
  _Float16 (*Kt)[64][72] = (_Float16 (*)[64][72])smemA;

  const int b    = blockIdx.x >> 4;
  const int qblk = blockIdx.x & 15;
  const int tid  = threadIdx.x;
  const int wave = tid >> 5;
  const int lane = tid & 31;
  const int lh   = lane & 15;
  const int hi   = lane >> 4;
  const int kb0  = hi * 8;
  const int qbase = qblk * 64 + wave * 16;

  const float*    qb  = q  + (size_t)b * C_ * WH;
  const _Float16* khb = kh + (size_t)b * LWH * 64;
  const _Float16* vhb = vh + (size_t)b * L_ * C_ * WH;

  // chunk decomposition for async staging: 512 x 16B per tensor, 4/thread
  const int prow = tid >> 1;                 // rows 0..63 (2 threads/row)
  const int pxo0 = (tid & 1) * 8;            // halfs; +16 per step (4 steps)

  // Q A-fragments (persist)
  v16h aq0, aq1;
  {
    const int row = qbase + lh;
    #pragma unroll
    for (int j = 0; j < 16; ++j) {
      const int cc = kb0 + j + ((j >= 8) ? 8 : 0);
      aq0[j] = (_Float16)qb[cc * WH + row];
      aq1[j] = (_Float16)qb[(cc + 32) * WH + row];
    }
  }

  v8f acc[4] = {};
  float mrow[8], lrow[8];
  #pragma unroll
  for (int r = 0; r < 8; ++r) { mrow[r] = -1e30f; lrow[r] = 0.f; }

  // ---- issue prefetch of tile 0 into buffer 0 ----
  {
    const _Float16* gk = khb;                            // [key][c], contiguous
    const _Float16* gv = vhb;                            // [c][key] rows stride 1024
    #pragma unroll
    for (int s = 0; s < 4; ++s) {
      const int xo = pxo0 + s * 16;
      async_ld16(gk + prow * 64 + xo,   &Kt[0][prow][xo]);
      async_ld16(gv + prow * 1024 + xo, &VtB[0][prow][xo]);
    }
  }

  for (int it = 0; it < 128; ++it) {
    const int buf = it & 1;
    wait_async0();
    __syncthreads();

    if (it + 1 < 128) {                       // prefetch next tile into buf^1
      const int ktn = (it + 1) * 64;
      const _Float16* gk = khb + (size_t)ktn * 64;
      const _Float16* gv = vhb + ((size_t)(ktn >> 10) * 64) * 1024 + (ktn & 1023);
      #pragma unroll
      for (int s = 0; s < 4; ++s) {
        const int xo = pxo0 + s * 16;
        async_ld16(gk + prow * 64 + xo,   &Kt[buf ^ 1][prow][xo]);
        async_ld16(gv + prow * 1024 + xo, &VtB[buf ^ 1][prow][xo]);
      }
    }

    _Float16 (*Ktc)[72] = Kt[buf];
    _Float16 (*Vtc)[72] = VtB[buf];

    // ---- scores ----
    v8f sc[4] = {};
    #pragma unroll
    for (int cf = 0; cf < 2; ++cf) {
      const v16h a = cf ? aq1 : aq0;
      #pragma unroll
      for (int t = 0; t < 4; ++t) {
        const v16h bk = ld16_split(&Ktc[t * 16 + lh][cf * 32 + hi * 8]);
        sc[t] = wmma_f16(a, bk, sc[t]);
      }
    }

    // ---- online softmax ----
    #pragma unroll
    for (int r = 0; r < 8; ++r) {
      float mx = fmaxf(fmaxf(sc[0][r], sc[1][r]), fmaxf(sc[2][r], sc[3][r]));
      mx = red_max16(mx);
      const float mnew  = fmaxf(mrow[r], mx);
      const float scale = __expf(mrow[r] - mnew);
      float rs = 0.f;
      #pragma unroll
      for (int t = 0; t < 4; ++t) {
        const float p = __expf(sc[t][r] - mnew);
        sc[t][r] = p; rs += p;
      }
      rs = red_sum16(rs);
      lrow[r] = lrow[r] * scale + rs;
      mrow[r] = mnew;
      #pragma unroll
      for (int t = 0; t < 4; ++t) acc[t][r] *= scale;
    }

    // ---- P: C-layout -> A-layout via wave-private LDS ----
    #pragma unroll
    for (int r = 0; r < 8; ++r)
      #pragma unroll
      for (int t = 0; t < 4; ++t)
        Pst[wave][r + hi * 8][t * 16 + lh] = (_Float16)sc[t][r];
    const v16h ap0 = ld16_split(&Pst[wave][lh][kb0]);
    const v16h ap1 = ld16_split(&Pst[wave][lh][32 + kb0]);

    // ---- acc += P @ V ----
    #pragma unroll
    for (int ct = 0; ct < 4; ++ct) {
      #pragma unroll
      for (int kf = 0; kf < 2; ++kf) {
        const v16h bv = ld16_split(&Vtc[ct * 16 + lh][kf * 32 + hi * 8]);
        acc[ct] = wmma_f16(kf ? ap1 : ap0, bv, acc[ct]);
      }
    }
  }

  // ---- epilogue: normalize, transpose via LDS (aliases Kt), write ----
  __syncthreads();
  float (*Ost)[16][65] = (float (*)[16][65])smemA;
  #pragma unroll
  for (int r = 0; r < 8; ++r) {
    const float inv = 1.0f / lrow[r];
    #pragma unroll
    for (int t = 0; t < 4; ++t)
      Ost[wave][r + hi * 8][t * 16 + lh] = acc[t][r] * inv;
  }
  __syncthreads();
  float* rsp = res_sp + (size_t)b * C_ * WH;
  for (int k = 0; k < 32; ++k) {
    const int flat = tid + k * 128;
    const int pixl = flat & 63;
    const int c    = flat >> 6;
    const int pix  = qblk * 64 + pixl;
    rsp[c * WH + pix] = Ost[pixl >> 4][pixl & 15][c] + qb[c * WH + pix];
  }
}

// ---------------------------------------------------------------------------
// Channel attention logits S[b][64][512] (K-dim = 1024 pixels).
// ---------------------------------------------------------------------------
__global__ __launch_bounds__(32) void chan_logits_kernel(
    const float* __restrict__ q, const float* __restrict__ keys,
    float* __restrict__ S)
{
  const int idx = blockIdx.x;
  const int b  = idx >> 7;
  const int mt = (idx >> 5) & 3;
  const int nt = idx & 31;
  const int lane = threadIdx.x;
  const int lh = lane & 15, hi = lane >> 4, kb0 = hi * 8;

  const float* qrow = q    + (size_t)b * C_ * WH + (size_t)(mt * 16 + lh) * WH;
  const float* krow = keys + (size_t)b * 512 * WH + (size_t)(nt * 16 + lh) * WH;

  v8f sacc = {};
  for (int kp = 0; kp < WH; kp += 32) {
    const float* ap = qrow + kp + kb0;
    const v16h a = cvt16(*(const float4*)ap, *(const float4*)(ap + 4),
                         *(const float4*)(ap + 16), *(const float4*)(ap + 20));
    const float* bp = krow + kp + hi * 8;
    const v16h bm = cvt16(*(const float4*)bp, *(const float4*)(bp + 4),
                          *(const float4*)(bp + 16), *(const float4*)(bp + 20));
    sacc = wmma_f16(a, bm, sacc);
  }
  float* Sb = S + (size_t)b * 64 * 512;
  #pragma unroll
  for (int r = 0; r < 8; ++r)
    Sb[(mt * 16 + r + hi * 8) * 512 + nt * 16 + lh] = sacc[r];
}

__global__ __launch_bounds__(256) void chan_softmax_kernel(float* __restrict__ S)
{
  float* Sr = S + (size_t)blockIdx.x * 512;
  const int t = threadIdx.x;
  __shared__ float red[256];
  const float a = Sr[t], b2 = Sr[t + 256];
  red[t] = fmaxf(a, b2); __syncthreads();
  for (int s = 128; s > 0; s >>= 1) { if (t < s) red[t] = fmaxf(red[t], red[t + s]); __syncthreads(); }
  const float m = red[0]; __syncthreads();
  const float e0 = __expf(a - m), e1 = __expf(b2 - m);
  red[t] = e0 + e1; __syncthreads();
  for (int s = 128; s > 0; s >>= 1) { if (t < s) red[t] += red[t + s]; __syncthreads(); }
  const float inv = 1.0f / red[0];
  Sr[t] = e0 * inv; Sr[t + 256] = e1 * inv;
}

__global__ __launch_bounds__(32) void chan_out_kernel(
    const float* __restrict__ P, const float* __restrict__ vals,
    const float* __restrict__ q, float* __restrict__ res_ch)
{
  const int idx = blockIdx.x;
  const int b  = idx >> 8;
  const int mt = (idx >> 6) & 3;
  const int nt = idx & 63;
  const int lane = threadIdx.x;
  const int lh = lane & 15, hi = lane >> 4, kb0 = hi * 8;

  const float* Prow = P + (size_t)b * 64 * 512 + (size_t)(mt * 16 + lh) * 512;
  const float* vb   = vals + (size_t)b * 512 * WH;
  const int colp = nt * 16 + lh;

  v8f acc = {};
  for (int kp = 0; kp < 512; kp += 32) {
    const float* ap = Prow + kp + kb0;
    const v16h a = cvt16(*(const float4*)ap, *(const float4*)(ap + 4),
                         *(const float4*)(ap + 16), *(const float4*)(ap + 20));
    v16h bv;
    #pragma unroll
    for (int j = 0; j < 16; ++j)
      bv[j] = (_Float16)vb[(size_t)(kp + hi * 8 + j + ((j >= 8) ? 8 : 0)) * WH + colp];
    acc = wmma_f16(a, bv, acc);
  }
  const float* qb = q + (size_t)b * C_ * WH;
  float* rb = res_ch + (size_t)b * C_ * WH;
  #pragma unroll
  for (int r = 0; r < 8; ++r) {
    const int cq = mt * 16 + r + hi * 8;
    rb[cq * WH + colp] = acc[r] + qb[cq * WH + colp];
  }
}

// ---------------------------------------------------------------------------
// 5x5 conv as implicit-GEMM WMMA (K = tap*64 + ci, tap-major).
// ---------------------------------------------------------------------------
__global__ __launch_bounds__(256) void wt_kernel(
    const float* __restrict__ w1, float* __restrict__ wt)
{
  const int i = blockIdx.x * 256 + threadIdx.x;
  if (i < 64 * 1600) {
    const int co = i / 1600, k = i % 1600;
    const int tap = k >> 6, ci = k & 63;
    wt[i] = w1[(size_t)(co * 64 + ci) * 25 + tap];
  }
}

__global__ __launch_bounds__(128) void conv5_wmma_kernel(
    const float* __restrict__ x, const float* __restrict__ wt,
    const float* __restrict__ b1, float* __restrict__ y)
{
  __shared__ __align__(32) _Float16 xp[6][36][72];
  const int b  = blockIdx.x >> 4;
  const int pt = blockIdx.x & 15;
  const int y0 = pt * 2;
  const int tid = threadIdx.x;
  const int wave = tid >> 5, lane = tid & 31;
  const int lh = lane & 15, hi = lane >> 4, kb0 = hi * 8;

  const float* xb = x + (size_t)b * C_ * WH;
  for (int i = tid; i < 64 * 6 * 36; i += 128) {
    const int ci = i / (6 * 36);
    const int rr = (i / 36) % 6;
    const int cc = i % 36;
    const int sy = y0 + rr - 2, sx = cc - 2;
    float v = 0.f;
    if (sy >= 0 && sy < 32 && sx >= 0 && sx < 32)
      v = xb[ci * WH + sy * 32 + sx];
    xp[rr][cc][ci] = (_Float16)v;
  }
  __syncthreads();

  const int lp = wave * 16 + lh;
  const int ly = lp >> 5, lx = lp & 31;

  v8f acc[4] = {};
  for (int s = 0; s < 50; ++s) {
    const int tap = s >> 1;
    const int dy = tap / 5, dx = tap % 5;
    const int cib = (s & 1) * 32;
    const v16h bf = ld16_split(&xp[ly + dy][lx + dx][cib + hi * 8]);
    const int kp = s * 32;
    #pragma unroll
    for (int mt = 0; mt < 4; ++mt) {
      const float* wr = wt + (size_t)(mt * 16 + lh) * 1600 + kp + kb0;
      const v16h af = cvt16(*(const float4*)wr, *(const float4*)(wr + 4),
                            *(const float4*)(wr + 16), *(const float4*)(wr + 20));
      acc[mt] = wmma_f16(af, bf, acc[mt]);
    }
  }
  float* yb = y + (size_t)b * C_ * WH + y0 * 32;
  #pragma unroll
  for (int mt = 0; mt < 4; ++mt)
    #pragma unroll
    for (int r = 0; r < 8; ++r) {
      const int co = mt * 16 + r + hi * 8;
      yb[co * WH + lp] = acc[mt][r] + b1[co];
    }
}

__global__ __launch_bounds__(256) void ln_stats_kernel(
    const float* __restrict__ y, float* __restrict__ stats)
{
  const int b = blockIdx.x;
  const float* yb = y + (size_t)b * C_ * WH;
  float s = 0.f, ss = 0.f;
  for (int i = threadIdx.x; i < C_ * WH; i += 256) {
    const float v = yb[i]; s += v; ss += v * v;
  }
  __shared__ float r1[256], r2[256];
  r1[threadIdx.x] = s; r2[threadIdx.x] = ss; __syncthreads();
  for (int o = 128; o > 0; o >>= 1) {
    if (threadIdx.x < o) { r1[threadIdx.x] += r1[threadIdx.x + o]; r2[threadIdx.x] += r2[threadIdx.x + o]; }
    __syncthreads();
  }
  if (threadIdx.x == 0) {
    const float n = (float)(C_ * WH);
    const float mu = r1[0] / n;
    const float var = r2[0] / n - mu * mu;
    stats[b * 2]     = mu;
    stats[b * 2 + 1] = rsqrtf(var + LN_EPS);
  }
}

__global__ __launch_bounds__(256) void ln_conv1_kernel(
    const float* __restrict__ y, const float* __restrict__ stats,
    const float* __restrict__ lnw, const float* __restrict__ lnb,
    const float* __restrict__ w2, const float* __restrict__ b2,
    float* __restrict__ out)
{
  const int b  = blockIdx.x >> 4;
  const int pt = blockIdx.x & 15;
  const int tid = threadIdx.x;
  __shared__ float zt[64][65];
  const float mu = stats[b * 2], rs = stats[b * 2 + 1];
  const float* yb = y + (size_t)b * C_ * WH;
  const int pbase = pt * 64;

  for (int k = 0; k < 16; ++k) {
    const int idx = tid + k * 256;
    const int pix = idx & 63, c = idx >> 6;
    const int gp = pbase + pix;
    const float v = (yb[c * WH + gp] - mu) * rs * lnw[c * WH + gp] + lnb[c * WH + gp];
    zt[c][pix] = fmaxf(v, 0.f);
  }
  __syncthreads();
  float* ob = out + (size_t)b * NH_ * WH;
  for (int k = 0; k < 32; ++k) {
    const int o = tid + k * 256;
    const int pix = o & 63, nh = o >> 6;
    float s = b2[nh];
    #pragma unroll 8
    for (int c = 0; c < 64; ++c)
      s += w2[nh * 64 + c] * zt[c][pix];
    ob[nh * WH + pbase + pix] = s;
  }
}

// ---------------------------------------------------------------------------
extern "C" void kernel_launch(void* const* d_in, const int* in_sizes, int n_in,
                              void* d_out, int out_size, void* d_ws, size_t ws_size,
                              hipStream_t stream) {
  const float* in_query = (const float*)d_in[0];
  const float* in_keys  = (const float*)d_in[1];
  const float* in_vals  = (const float*)d_in[2];
  const float* s_w1 = (const float*)d_in[3];
  const float* s_b1 = (const float*)d_in[4];
  const float* s_lnw = (const float*)d_in[5];
  const float* s_lnb = (const float*)d_in[6];
  const float* s_w2 = (const float*)d_in[7];
  const float* s_b2 = (const float*)d_in[8];
  const float* c_w1 = (const float*)d_in[9];
  const float* c_b1 = (const float*)d_in[10];
  const float* c_lnw = (const float*)d_in[11];
  const float* c_lnb = (const float*)d_in[12];
  const float* c_w2 = (const float*)d_in[13];
  const float* c_b2 = (const float*)d_in[14];

  float* ws = (float*)d_ws;
  const size_t PLANE = (size_t)B_ * C_ * WH;           // 524288
  float* res_sp  = ws;
  float* res_ch  = ws + PLANE;
  float* S_ch    = ws + 2 * PLANE;                     // 262144
  float* y_s     = ws + 2 * PLANE + 262144;
  float* y_c     = y_s + PLANE;
  float* stats_s = y_c + PLANE;
  float* stats_c = stats_s + 16;
  float* wt_s    = stats_c + 16;                       // 64*1600
  float* wt_c    = wt_s + 64 * 1600;
  _Float16* kh   = (_Float16*)(wt_c + 64 * 1600);      // B*LWH*64 halfs
  _Float16* vh   = kh + (size_t)B_ * LWH * 64;

  float* out_s = (float*)d_out;
  float* out_c = out_s + (size_t)B_ * NH_ * WH;

  // f16 pre-conversion for async-DMA attention tiles
  khconv_kernel<<<B_ * LWH / 256, 256, 0, stream>>>(in_keys, kh);
  vhconv_kernel<<<B_ * LWH * 64 / 1024, 256, 0, stream>>>(in_vals, vh);
  // weight pre-transpose for implicit-GEMM conv
  wt_kernel<<<400, 256, 0, stream>>>(s_w1, wt_s);
  wt_kernel<<<400, 256, 0, stream>>>(c_w1, wt_c);
  // spatial branch attention (async double-buffered)
  spatial_attn_kernel<<<B_ * 16, 128, 0, stream>>>(in_query, kh, vh, res_sp);
  // channel branch attention
  chan_logits_kernel<<<B_ * 4 * 32, 32, 0, stream>>>(in_query, in_keys, S_ch);
  chan_softmax_kernel<<<B_ * 64, 256, 0, stream>>>(S_ch);
  chan_out_kernel<<<B_ * 4 * 64, 32, 0, stream>>>(S_ch, in_vals, in_query, res_ch);
  // stems
  conv5_wmma_kernel<<<B_ * 16, 128, 0, stream>>>(res_sp, wt_s, s_b1, y_s);
  conv5_wmma_kernel<<<B_ * 16, 128, 0, stream>>>(res_ch, wt_c, c_b1, y_c);
  ln_stats_kernel<<<B_, 256, 0, stream>>>(y_s, stats_s);
  ln_stats_kernel<<<B_, 256, 0, stream>>>(y_c, stats_c);
  ln_conv1_kernel<<<B_ * 16, 256, 0, stream>>>(y_s, stats_s, s_lnw, s_lnb, s_w2, s_b2, out_s);
  ln_conv1_kernel<<<B_ * 16, 256, 0, stream>>>(y_c, stats_c, c_lnw, c_lnb, c_w2, c_b2, out_c);
}
